// STNet_model_83150566851115
// MI455X (gfx1250) — matrix operand order repaired and checked by reading, not compiled, not run on hardware
//
#include <hip/hip_runtime.h>
#include <hip/hip_bf16.h>
#include <math.h>

// ---------------------------------------------------------------------------
// Model dims
// ---------------------------------------------------------------------------
#define BSZ     256            // batch
#define NP_     50             // segments per sample (T)
#define PS_     1024           // samples per segment
#define NS_     64             // fft size
#define NNODES  33             // NS/2+1
#define NFR     17             // frames per segment
#define ROWS    (BSZ*NP_)      // 12800
#define D0      (NNODES*64)    // 2112 encoder input
#define D1      1024           // hidden
#define LH_     512            // lstm hidden
#define G4      2048           // 4*LH

typedef __attribute__((ext_vector_type(16))) __bf16 v16bf;
typedef __attribute__((ext_vector_type(8)))  float  v8f;

static __device__ __forceinline__ unsigned short f2bf(float f) {
    union { float f; unsigned u; } x; x.f = f;
    unsigned r = x.u + 0x7FFFu + ((x.u >> 16) & 1u);   // RNE
    return (unsigned short)(r >> 16);
}
static __device__ __forceinline__ float sigf(float x) {
    return 1.0f / (1.0f + __expf(-x));
}

// ---------------------------------------------------------------------------
// Weight conversion kernels
// ---------------------------------------------------------------------------
__global__ void k_cvt_bf16(const float* __restrict__ s, unsigned short* __restrict__ d, int n) {
    int i = blockIdx.x * 256 + threadIdx.x;
    if (i < n) d[i] = f2bf(s[i]);
}
// src is R x C row-major; dst is C x R row-major (transpose)
__global__ void k_cvtT_bf16(const float* __restrict__ s, unsigned short* __restrict__ d, int R, int C) {
    int i = blockIdx.x * 256 + threadIdx.x;
    if (i < R * C) {
        int r = i / C, c = i % C;
        d[(size_t)c * R + r] = f2bf(s[i]);
    }
}
__global__ void k_bsum(const float* __restrict__ a, const float* __restrict__ b,
                       float* __restrict__ o, int n) {
    int i = blockIdx.x * 256 + threadIdx.x;
    if (i < n) o[i] = a[i] + b[i];
}

// ---------------------------------------------------------------------------
// STFT magnitude + adjacency vector h (rank-1 graph) per segment
// ---------------------------------------------------------------------------
__global__ __launch_bounds__(256) void k_stft(
    const float* __restrict__ x, const float* __restrict__ cnn_w,
    const float* __restrict__ cnn_b,
    float* __restrict__ xg, float* __restrict__ hvec, float* __restrict__ cnt)
{
    const int g = blockIdx.x;               // segment id 0..ROWS-1
    const int tid = threadIdx.x;
    __shared__ float sx[PS_];
    __shared__ float wf[NFR * NS_];
    __shared__ float mg[NNODES * NFR];
    __shared__ float c64[NS_], s64[NS_];
    __shared__ float hsh[NNODES];

    const float* xs = x + (size_t)g * PS_;
    for (int i = tid; i < PS_; i += 256) sx[i] = xs[i];
    if (tid < NS_) {
        float ang = 6.28318530717958647692f * (float)tid / (float)NS_;
        c64[tid] = __cosf(ang);
        s64[tid] = __sinf(ang);
    }
    __syncthreads();

    // reflect-padded, Hann-windowed frames (hop == NS)
    for (int i = tid; i < NFR * NS_; i += 256) {
        int t = i >> 6, k = i & 63;
        int j = t * NS_ + k - 32;
        if (j < 0) j = -j;
        if (j > PS_ - 1) j = 2 * (PS_ - 1) - j;
        float w = 0.5f * (1.0f - c64[k]);   // 0.5*(1-cos(2πk/64))
        wf[i] = sx[j] * w;
    }
    __syncthreads();

    // 33-bin rDFT magnitude per frame, output transposed to [node][frame]
    for (int p = tid; p < NNODES * NFR; p += 256) {
        int t = p % NFR, m = p / NFR;
        const float* v = &wf[t * NS_];
        float re = 0.f, im = 0.f;
        #pragma unroll 8
        for (int k = 0; k < NS_; ++k) {
            int a = (m * k) & 63;
            re += v[k] * c64[a];
            im -= v[k] * s64[a];
        }
        float mag = sqrtf(re * re + im * im);
        mg[m * NFR + t] = mag;
        xg[(size_t)g * (NNODES * NFR) + m * NFR + t] = mag;
    }
    __syncthreads();

    // feats = [mean, max] over frames; nw = feats @ cnn_w + b; h = nw > 0.7
    if (tid < NNODES) {
        float s = 0.f, mx = 0.f;
        for (int t = 0; t < NFR; ++t) {
            float v = mg[tid * NFR + t];
            s += v; mx = fmaxf(mx, v);
        }
        float nw = (s / (float)NFR) * cnn_w[0] + mx * cnn_w[1] + cnn_b[0];
        float hv = (nw > 0.7f) ? 1.0f : 0.0f;
        hsh[tid] = hv;
        hvec[(size_t)g * NNODES + tid] = hv;
    }
    __syncthreads();
    if (tid == 0) {
        float c = 0.f;
        for (int n = 0; n < NNODES; ++n) c += hsh[n];
        cnt[g] = c;
    }
}

// ---------------------------------------------------------------------------
// Chebyshev graph conv (x2) exploiting rank-1 adjacency:
//   adj@X = h ⊗ (hᵀX);  Tx2 = 2c·h⊗s − Tx0  (c = Σh since h binary)
//   cheb(X,F) = X@(F0−F2) + h ⊗ (s@F1 + 2c·s@F2),  s = hᵀX
// ---------------------------------------------------------------------------
__global__ __launch_bounds__(256) void k_cheb(
    const float* __restrict__ xg, const float* __restrict__ hvec,
    const float* __restrict__ cnt, const float* __restrict__ cheb0,
    const float* __restrict__ cheb1, unsigned short* __restrict__ Yo)
{
    const int g = blockIdx.x;
    const int tid = threadIdx.x;
    __shared__ float xc[NNODES * NFR];
    __shared__ float hh[NNODES];
    __shared__ float s0[NFR], u1[64];
    __shared__ float G0[NFR * 64];      // cheb0[0]-cheb0[2]
    __shared__ float y1[NNODES * 64];
    __shared__ float s1[64], u2[64];
    __shared__ float G2[64 * 64];       // cheb1[0]-cheb1[2]
    __shared__ float cl;

    for (int i = tid; i < NNODES * NFR; i += 256) xc[i] = xg[(size_t)g * (NNODES * NFR) + i];
    if (tid < NNODES) hh[tid] = hvec[(size_t)g * NNODES + tid];
    if (tid == 0) cl = cnt[g];
    for (int i = tid; i < NFR * 64; i += 256) G0[i] = cheb0[i] - cheb0[2 * NFR * 64 + i];
    __syncthreads();

    if (tid < NFR) {                           // s0 = hᵀ xc
        float s = 0.f;
        for (int n = 0; n < NNODES; ++n) s += hh[n] * xc[n * NFR + tid];
        s0[tid] = s;
    }
    __syncthreads();
    if (tid < 64) {                            // u1 = s0@F1 + 2c·s0@F2
        float a = 0.f, b = 0.f;
        for (int k = 0; k < NFR; ++k) {
            a += s0[k] * cheb0[NFR * 64 + k * 64 + tid];
            b += s0[k] * cheb0[2 * NFR * 64 + k * 64 + tid];
        }
        u1[tid] = a + 2.0f * cl * b;
    }
    __syncthreads();
    for (int p = tid; p < NNODES * 64; p += 256) {   // y1 = xc@G0 + h⊗u1
        int n = p >> 6, j = p & 63;
        float a = 0.f;
        #pragma unroll
        for (int k = 0; k < NFR; ++k) a += xc[n * NFR + k] * G0[k * 64 + j];
        y1[p] = a + hh[n] * u1[j];
    }
    for (int i = tid; i < 64 * 64; i += 256) G2[i] = cheb1[i] - cheb1[2 * 64 * 64 + i];
    __syncthreads();

    if (tid < 64) {                            // s1 = hᵀ y1
        float s = 0.f;
        for (int n = 0; n < NNODES; ++n) s += hh[n] * y1[n * 64 + tid];
        s1[tid] = s;
    }
    __syncthreads();
    if (tid < 64) {                            // u2 = s1@F1' + 2c·s1@F2'
        float a = 0.f, b = 0.f;
        for (int k = 0; k < 64; ++k) {
            a += s1[k] * cheb1[64 * 64 + k * 64 + tid];
            b += s1[k] * cheb1[2 * 64 * 64 + k * 64 + tid];
        }
        u2[tid] = a + 2.0f * cl * b;
    }
    __syncthreads();
    for (int p = tid; p < NNODES * 64; p += 256) {   // y2 = y1@G2 + h⊗u2 -> bf16
        int n = p >> 6, j = p & 63;
        float a = 0.f;
        #pragma unroll 8
        for (int k = 0; k < 64; ++k) a += y1[n * 64 + k] * G2[k * 64 + j];
        Yo[(size_t)g * D0 + p] = f2bf(a + hh[n] * u2[j]);
    }
}

// ---------------------------------------------------------------------------
// bf16 WMMA GEMM core. Block 256 thr (8 waves), tile BM=128 x BN=128, BK=32.
// A tile in LDS row-major [128][32+8]; B tile in LDS TRANSPOSED (N-major)
// [128][32+8] so both A and B fragments are 2x ds_load_b128 each
// (CDNA5 ISA 7.12.2 wave32 layouts). Per K-step each wave:
//   - prefetches the next K-step's global tiles into registers (overlaps WMMA)
//   - loads 1 A fragment + all 8 B fragments, then 8 back-to-back WMMAs
// Requires M%128==0, N%128==0, K%32==0 (true for all call sites).
// ---------------------------------------------------------------------------
#define BM 128
#define BN 128
#define BK 32
#define AP 40   // padded K-stride for A tile rows (ushorts); 80B, 16B-aligned
#define BP 40   // padded K-stride for transposed B tile rows (ushorts)

static __device__ __forceinline__ void gemm_core(
    const unsigned short* __restrict__ A, const unsigned short* __restrict__ B,
    int N, int K, int m0, int n0,
    unsigned short* As, unsigned short* Bs, v8f acc[8])
{
    const int tid  = threadIdx.x;
    const int lane = tid & 31;
    const int wave = tid >> 5;
    const int hl   = lane >> 4;
    const int l16  = lane & 15;
    const int arow = tid >> 1, aseg = (tid & 1) * 16;   // A: 2 thr/row, 16 elems
    const int brow = tid >> 3, bcol = (tid & 7) * 16;   // B: 8 thr/row, 16 elems

    const uint4* ga = (const uint4*)(A + (size_t)(m0 + arow) * K + aseg);
    const unsigned short* gb = B + (size_t)brow * N + n0 + bcol;

    uint4 a0 = ga[0], a1 = ga[1];
    union { uint4 q[2]; unsigned short u[16]; } bv;
    bv.q[0] = ((const uint4*)gb)[0];
    bv.q[1] = ((const uint4*)gb)[1];

    for (int k0 = 0; k0 < K; k0 += BK) {
        // stage current tiles to LDS
        *(uint4*)&As[arow * AP + aseg]     = a0;
        *(uint4*)&As[arow * AP + aseg + 8] = a1;
        #pragma unroll
        for (int i = 0; i < 16; ++i)                    // transpose: Bs[n][k]
            Bs[(bcol + i) * BP + brow] = bv.u[i];
        __syncthreads();

        // prefetch next K-step's tiles (overlaps with WMMA below)
        if (k0 + BK < K) {
            const uint4* gan = (const uint4*)(A + (size_t)(m0 + arow) * K + (k0 + BK) + aseg);
            a0 = gan[0]; a1 = gan[1];
            const uint4* gbn = (const uint4*)(B + (size_t)(k0 + BK + brow) * N + n0 + bcol);
            bv.q[0] = gbn[0]; bv.q[1] = gbn[1];
        }

        // A fragment (CDNA5 16-bit A 16x32 layout)
        union { v16bf v; uint4 q[2]; } af;
        const int mr = wave * 16 + l16;
        af.q[0] = *(const uint4*)&As[mr * AP + hl * 8];       // K = hl*8..+7
        af.q[1] = *(const uint4*)&As[mr * AP + 16 + hl * 8];  // K = 16+hl*8..+7

        // all 8 B fragments first, then 8 independent WMMAs back-to-back
        union { v16bf v; uint4 q[2]; } bfr[8];
        #pragma unroll
        for (int nt = 0; nt < 8; ++nt) {
            const int nn = nt * 16 + l16;
            bfr[nt].q[0] = *(const uint4*)&Bs[nn * BP + hl * 8];
            bfr[nt].q[1] = *(const uint4*)&Bs[nn * BP + 16 + hl * 8];
        }
        #pragma unroll
        for (int nt = 0; nt < 8; ++nt)
            acc[nt] = __builtin_amdgcn_wmma_f32_16x16x32_bf16(
                false, af.v, false, bfr[nt].v, (short)0, acc[nt], false, false);

        __syncthreads();
    }
}

// OUTF32: write fp32 output; else bf16 output. RELU: apply max(0,.).
template <bool OUTF32, bool RELU>
__global__ __launch_bounds__(256) void k_wmma_gemm(
    const unsigned short* __restrict__ A, const unsigned short* __restrict__ B,
    const float* __restrict__ bias, void* __restrict__ Cout,
    int N, int K)
{
    __shared__ unsigned short As[BM * AP];
    __shared__ unsigned short Bs[BN * BP];
    const int tid = threadIdx.x;
    const int lane = tid & 31, wave = tid >> 5;
    const int hl = lane >> 4, l16 = lane & 15;
    const int m0 = blockIdx.y * BM, n0 = blockIdx.x * BN;

    const v8f vz = {0.f, 0.f, 0.f, 0.f, 0.f, 0.f, 0.f, 0.f};
    v8f acc[8] = {vz, vz, vz, vz, vz, vz, vz, vz};
    gemm_core(A, B, N, K, m0, n0, As, Bs, acc);

    const int mrow = m0 + wave * 16 + hl * 8;   // C/D: M = 8*(lane>>4)+r
    #pragma unroll
    for (int nt = 0; nt < 8; ++nt) {
        const int gn = n0 + nt * 16 + l16;
        const float bvv = bias[gn];
        #pragma unroll
        for (int r = 0; r < 8; ++r) {
            float v = acc[nt][r] + bvv;
            if (RELU) v = fmaxf(v, 0.f);
            const size_t off = (size_t)(mrow + r) * N + gn;
            if (OUTF32) ((float*)Cout)[off] = v;
            else        ((unsigned short*)Cout)[off] = f2bf(v);
        }
    }
}

// LSTM recurrent GEMM: gates(256x2048) = gates_x[:,t] + h_prev(256x512)@WhhT
// gates_x rows are laid out (b*50 + t).
__global__ __launch_bounds__(256) void k_lstm_gemm(
    const unsigned short* __restrict__ A,   // h_prev bf16 256x512
    const unsigned short* __restrict__ B,   // WhhT bf16 512x2048
    const float* __restrict__ gates_x,      // 12800x2048
    float* __restrict__ gates,              // 256x2048
    int t)
{
    __shared__ unsigned short As[BM * AP];
    __shared__ unsigned short Bs[BN * BP];
    const int tid = threadIdx.x;
    const int lane = tid & 31, wave = tid >> 5;
    const int hl = lane >> 4, l16 = lane & 15;
    const int m0 = blockIdx.y * BM, n0 = blockIdx.x * BN;

    const v8f vz = {0.f, 0.f, 0.f, 0.f, 0.f, 0.f, 0.f, 0.f};
    v8f acc[8] = {vz, vz, vz, vz, vz, vz, vz, vz};
    gemm_core(A, B, G4, LH_, m0, n0, As, Bs, acc);

    const int mrow = m0 + wave * 16 + hl * 8;
    #pragma unroll
    for (int nt = 0; nt < 8; ++nt) {
        const int gn = n0 + nt * 16 + l16;
        #pragma unroll
        for (int r = 0; r < 8; ++r) {
            const int b = mrow + r;
            float v = acc[nt][r] + gates_x[(size_t)(b * NP_ + t) * G4 + gn];
            gates[(size_t)b * G4 + gn] = v;
        }
    }
}

// Elementwise LSTM cell update
__global__ void k_lstm_cell(const float* __restrict__ gates,
                            float* __restrict__ cbuf,
                            unsigned short* __restrict__ hbf,
                            float* __restrict__ lstm_out, int t)
{
    int idx = blockIdx.x * 256 + threadIdx.x;      // 256*512 threads
    if (idx >= BSZ * LH_) return;
    int b = idx >> 9, j = idx & (LH_ - 1);
    const float* gr = gates + (size_t)b * G4;
    float ig = sigf(gr[j]);
    float fg = sigf(gr[LH_ + j]);
    float gv = tanhf(gr[2 * LH_ + j]);
    float og = sigf(gr[3 * LH_ + j]);
    float c = fg * cbuf[idx] + ig * gv;
    float h = og * tanhf(c);
    cbuf[idx] = c;
    hbf[idx] = f2bf(h);
    lstm_out[(size_t)b * (NP_ * LH_) + t * LH_ + j] = h;
}

// out[b] = lstm_out[b,:] . out_w + out_b
__global__ __launch_bounds__(256) void k_final(
    const float* __restrict__ lstm_out, const float* __restrict__ out_w,
    const float* __restrict__ out_b, float* __restrict__ out)
{
    __shared__ float red[256];
    const int b = blockIdx.x, tid = threadIdx.x;
    float s = 0.f;
    const float* row = lstm_out + (size_t)b * (NP_ * LH_);
    for (int i = tid; i < NP_ * LH_; i += 256) s += row[i] * out_w[i];
    red[tid] = s;
    __syncthreads();
    for (int st = 128; st > 0; st >>= 1) {
        if (tid < st) red[tid] += red[tid + st];
        __syncthreads();
    }
    if (tid == 0) out[b] = red[0] + out_b[0];
}

// ---------------------------------------------------------------------------
// Host orchestration
// ---------------------------------------------------------------------------
static inline size_t al256(size_t x) { return (x + 255) & ~(size_t)255; }

extern "C" void kernel_launch(void* const* d_in, const int* in_sizes, int n_in,
                              void* d_out, int out_size, void* d_ws, size_t ws_size,
                              hipStream_t stream) {
    const float* x      = (const float*)d_in[0];
    const float* cnn_w  = (const float*)d_in[1];
    const float* cnn_b  = (const float*)d_in[2];
    const float* cheb0  = (const float*)d_in[3];
    const float* cheb1  = (const float*)d_in[4];
    const float* enc_w0 = (const float*)d_in[5];
    const float* enc_b0 = (const float*)d_in[6];
    const float* enc_w1 = (const float*)d_in[7];
    const float* enc_b1 = (const float*)d_in[8];
    const float* enc_w2 = (const float*)d_in[9];
    const float* enc_b2 = (const float*)d_in[10];
    const float* enc_w3 = (const float*)d_in[11];
    const float* enc_b3 = (const float*)d_in[12];
    const float* W_ih   = (const float*)d_in[13];
    const float* W_hh   = (const float*)d_in[14];
    const float* b_ih   = (const float*)d_in[15];
    const float* b_hh   = (const float*)d_in[16];
    const float* out_w  = (const float*)d_in[17];
    const float* out_b  = (const float*)d_in[18];
    float* out = (float*)d_out;
    (void)in_sizes; (void)n_in; (void)out_size; (void)ws_size;

    char* w = (char*)d_ws;
    size_t off = 0;
    auto alloc = [&](size_t bytes) { void* p = w + off; off += al256(bytes); return p; };

    float*          xg     = (float*)alloc((size_t)ROWS * NNODES * NFR * 4);
    float*          hvec   = (float*)alloc((size_t)ROWS * NNODES * 4);
    float*          cntb   = (float*)alloc((size_t)ROWS * 4);
    unsigned short* Yo     = (unsigned short*)alloc((size_t)ROWS * D0 * 2);
    unsigned short* w0b    = (unsigned short*)alloc((size_t)D0 * D1 * 2);
    unsigned short* w1b    = (unsigned short*)alloc((size_t)D1 * D1 * 2);
    unsigned short* w2b    = (unsigned short*)alloc((size_t)D1 * D1 * 2);
    unsigned short* w3b    = (unsigned short*)alloc((size_t)D1 * D1 * 2);
    unsigned short* WihT   = (unsigned short*)alloc((size_t)D1 * G4 * 2);
    unsigned short* WhhT   = (unsigned short*)alloc((size_t)LH_ * G4 * 2);
    float*          bsum   = (float*)alloc((size_t)G4 * 4);
    unsigned short* Hb1    = (unsigned short*)alloc((size_t)ROWS * D1 * 2);
    unsigned short* Hb2    = (unsigned short*)alloc((size_t)ROWS * D1 * 2);
    float*          gatesx = (float*)alloc((size_t)ROWS * G4 * 4);
    float*          gatest = (float*)alloc((size_t)BSZ * G4 * 4);
    unsigned short* hbf    = (unsigned short*)alloc((size_t)BSZ * LH_ * 2);
    float*          cbuf   = (float*)alloc((size_t)BSZ * LH_ * 4);
    float*          lstm   = (float*)alloc((size_t)BSZ * NP_ * LH_ * 4);

    auto cvt = [&](const float* s, unsigned short* d, int n) {
        k_cvt_bf16<<<(n + 255) / 256, 256, 0, stream>>>(s, d, n);
    };
    cvt(enc_w0, w0b, D0 * D1);
    cvt(enc_w1, w1b, D1 * D1);
    cvt(enc_w2, w2b, D1 * D1);
    cvt(enc_w3, w3b, D1 * D1);
    k_cvtT_bf16<<<(G4 * D1 + 255) / 256, 256, 0, stream>>>(W_ih, WihT, G4, D1);
    k_cvtT_bf16<<<(G4 * LH_ + 255) / 256, 256, 0, stream>>>(W_hh, WhhT, G4, LH_);
    k_bsum<<<(G4 + 255) / 256, 256, 0, stream>>>(b_ih, b_hh, bsum, G4);

    k_stft<<<ROWS, 256, 0, stream>>>(x, cnn_w, cnn_b, xg, hvec, cntb);
    k_cheb<<<ROWS, 256, 0, stream>>>(xg, hvec, cntb, cheb0, cheb1, Yo);

    // Encoder: Yo -> H1 -> H2 -> H1 -> H2 (final H), then gates_x = H@WihT + bsum
    k_wmma_gemm<false, true><<<dim3(D1 / BN, ROWS / BM), 256, 0, stream>>>(
        Yo, w0b, enc_b0, Hb1, D1, D0);
    k_wmma_gemm<false, true><<<dim3(D1 / BN, ROWS / BM), 256, 0, stream>>>(
        Hb1, w1b, enc_b1, Hb2, D1, D1);
    k_wmma_gemm<false, true><<<dim3(D1 / BN, ROWS / BM), 256, 0, stream>>>(
        Hb2, w2b, enc_b2, Hb1, D1, D1);
    k_wmma_gemm<false, false><<<dim3(D1 / BN, ROWS / BM), 256, 0, stream>>>(
        Hb1, w3b, enc_b3, Hb2, D1, D1);
    k_wmma_gemm<true, false><<<dim3(G4 / BN, ROWS / BM), 256, 0, stream>>>(
        Hb2, WihT, bsum, gatesx, G4, D1);

    // LSTM: h0 = c0 = 0
    hipMemsetAsync(hbf, 0, (size_t)BSZ * LH_ * 2, stream);
    hipMemsetAsync(cbuf, 0, (size_t)BSZ * LH_ * 4, stream);
    for (int t = 0; t < NP_; ++t) {
        k_lstm_gemm<<<dim3(G4 / BN, BSZ / BM), 256, 0, stream>>>(
            hbf, WhhT, gatesx, gatest, t);
        k_lstm_cell<<<(BSZ * LH_) / 256, 256, 0, stream>>>(
            gatest, cbuf, hbf, lstm, t);
    }

    k_final<<<BSZ, 256, 0, stream>>>(lstm, out_w, out_b, out);
}
